// graph_net_66632122630565
// MI455X (gfx1250) — matrix operand order repaired and checked by reading, not compiled, not run on hardware
//
#include <hip/hip_runtime.h>
#include <hip/hip_bf16.h>

#define N_NODES   16384
#define HID_IN    1024
#define HID_MID   1500
#define HID_MID_P 1536   // padded to multiple of 64 for tiling (also K of GEMM2)

typedef __attribute__((ext_vector_type(16))) __bf16 v16bf;
typedef __attribute__((ext_vector_type(8)))  __bf16 v8bf;
typedef __attribute__((ext_vector_type(8)))  float  v8f;

// GCC-style vector to match the async-to-LDS builtin's parameter type
typedef int v4i_vs __attribute__((vector_size(16)));
typedef __attribute__((address_space(1))) v4i_vs* as1_v4i_ptr;
typedef __attribute__((address_space(3))) v4i_vs* as3_v4i_ptr;

#if __has_builtin(__builtin_amdgcn_global_load_async_to_lds_b128)
#define HAVE_ASYNC_LDS 1
#else
#define HAVE_ASYNC_LDS 0
#endif

// ---------------------------------------------------------------- utilities

__device__ __forceinline__ void split_bf16(float x, __bf16& hi, __bf16& lo) {
    __bf16 h = (__bf16)x;             // RNE
    hi = h;
    lo = (__bf16)(x - (float)h);      // residual captures next ~8 mantissa bits
}

// 16-byte global -> LDS copy; async (ASYNCcnt-tracked) when available.
__device__ __forceinline__ void g2l_b128(__bf16* l, const __bf16* g) {
#if HAVE_ASYNC_LDS
    __builtin_amdgcn_global_load_async_to_lds_b128(
        (as1_v4i_ptr)(unsigned long long)g,
        (as3_v4i_ptr)(unsigned)(unsigned long long)l,
        0, 0);
#else
    *(v8bf*)l = *(const v8bf*)g;      // sync fallback: global load + ds_store
#endif
}

__device__ __forceinline__ void wait_async_le2() {
#if HAVE_ASYNC_LDS
#if __has_builtin(__builtin_amdgcn_s_wait_asynccnt)
    __builtin_amdgcn_s_wait_asynccnt(2);
#else
    asm volatile("s_wait_asynccnt 0x2" ::: "memory");
#endif
#endif
}

__device__ __forceinline__ void wait_async_le0() {
#if HAVE_ASYNC_LDS
#if __has_builtin(__builtin_amdgcn_s_wait_asynccnt)
    __builtin_amdgcn_s_wait_asynccnt(0);
#else
    asm volatile("s_wait_asynccnt 0x0" ::: "memory");
#endif
#endif
}

// ------------------------------------------------------------ degree / norm

__global__ void deg_init_kernel(float* __restrict__ isd) {
    int i = blockIdx.x * blockDim.x + threadIdx.x;
    if (i < N_NODES) isd[i] = 1.0f;   // self-loop contributes 1 to degree
}

__global__ void deg_count_kernel(const long long* __restrict__ dst,
                                 float* __restrict__ isd, int nedges) {
    int e = blockIdx.x * blockDim.x + threadIdx.x;
    if (e < nedges) atomicAdd(&isd[(int)dst[e]], 1.0f);
}

__global__ void deg_final_kernel(float* __restrict__ isd) {
    int i = blockIdx.x * blockDim.x + threadIdx.x;
    if (i < N_NODES) isd[i] = rsqrtf(isd[i]);   // deg >= 1 always (self-loop)
}

// ------------------------------------------------------ fp32 -> bf16 splits

__global__ void split_mat_kernel(const float* __restrict__ X,
                                 __bf16* __restrict__ hi,
                                 __bf16* __restrict__ lo, size_t total) {
    size_t idx = (size_t)blockIdx.x * blockDim.x + threadIdx.x;
    if (idx >= total) return;
    split_bf16(X[idx], hi[idx], lo[idx]);
}

// W [Kin x Nin] row-major -> WT [rowsOut x colsOut] row-major (transposed,
// zero-padded): WT[r][c] = (r<Nin && c<Kin) ? W[c][r] : 0
__global__ void transpose_split_pad_kernel(const float* __restrict__ W,
                                           __bf16* __restrict__ Thi,
                                           __bf16* __restrict__ Tlo,
                                           int Kin, int Nin,
                                           int rowsOut, int colsOut) {
    size_t idx = (size_t)blockIdx.x * blockDim.x + threadIdx.x;
    size_t total = (size_t)rowsOut * colsOut;
    if (idx >= total) return;
    int rr = (int)(idx / colsOut);
    int cc = (int)(idx % colsOut);
    float v = (rr < Nin && cc < Kin) ? W[(size_t)cc * Nin + rr] : 0.0f;
    split_bf16(v, Thi[idx], Tlo[idx]);
}

// ------------------------------------------------------------- WMMA GEMM
// C[M x Npad] = A[M x K] * B[K x Npad].  A: split bf16 row-major.
// B: split bf16 TRANSPOSED row-major (BT[Npad x K]).
// Block = 8 waves = 128(M) x 64(N); the 64-column B slab is staged in LDS
// (double-buffered, async-to-LDS) and shared by all 8 waves.  Each wave owns a
// 16x64 tile; 3 bf16 WMMAs per (K=32, 16-wide-N) chunk emulate fp32 precision:
// Ahi*Bhi + Ahi*Blo + Alo*Bhi.  Requires M%128==0, Npad%64==0, K%32==0.

__global__ void __launch_bounds__(256)
gemm_bf16split_lds_kernel(const __bf16* __restrict__ Ahi,
                          const __bf16* __restrict__ Alo,
                          const __bf16* __restrict__ BThi,
                          const __bf16* __restrict__ BTlo,
                          float* __restrict__ C,
                          int M, int K, int Npad) {
    // [buf][col*32 + kk] ; 64 cols x 32 k of bf16 = 4KB per plane
    __shared__ __align__(64) __bf16 BsHi[2][64 * 32];
    __shared__ __align__(64) __bf16 BsLo[2][64 * 32];

    const int ngc = Npad >> 6;
    const int mb  = (int)blockIdx.x / ngc;
    const int nb  = (int)blockIdx.x % ngc;
    const int n0  = nb * 64;

    const int tid  = threadIdx.x;
    const int wave = tid >> 5;
    const int lane = tid & 31;
    const int r    = lane & 15;
    const int hs   = lane >> 4;
    const int row0 = mb * 128 + wave * 16;

    // B staging: 256 threads, each copies one 16B chunk of hi and one of lo.
    const int bc = tid >> 2;                       // column 0..63
    const int bq = tid & 3;                        // 16B chunk within column
    const __bf16* gHi = BThi + (size_t)(n0 + bc) * K + bq * 8;
    const __bf16* gLo = BTlo + (size_t)(n0 + bc) * K + bq * 8;
    const int stOff   = bc * 32 + bq * 8;          // element offset in plane

    const __bf16* aHiRow = Ahi + (size_t)(row0 + r) * K;
    const __bf16* aLoRow = Alo + (size_t)(row0 + r) * K;

    v8f acc[4] = {v8f{}, v8f{}, v8f{}, v8f{}};

    // prologue: stage k0 = 0 into buffer 0
    g2l_b128(&BsHi[0][stOff], gHi);
    g2l_b128(&BsLo[0][stOff], gLo);

    const int iters = K >> 5;
    for (int it = 0; it < iters; ++it) {
        const int buf = it & 1;
        const int k0  = it << 5;

        if (it + 1 < iters) {           // prefetch next slab into other buffer
            g2l_b128(&BsHi[buf ^ 1][stOff], gHi + k0 + 32);
            g2l_b128(&BsLo[buf ^ 1][stOff], gLo + k0 + 32);
            wait_async_le2();           // in-order: current buffer's 2 are done
        } else {
            wait_async_le0();
        }
        __syncthreads();                // all waves' staged data visible

        // A fragment: lanes 0-15 hold K {0..15}, lanes 16-31 K {16..31}
        const int ka = k0 + hs * 16;
        v8bf a0 = *(const v8bf*)(aHiRow + ka);
        v8bf a1 = *(const v8bf*)(aHiRow + ka + 8);
        v16bf ahi = __builtin_shufflevector(a0, a1,
            0,1,2,3,4,5,6,7,8,9,10,11,12,13,14,15);
        v8bf l0 = *(const v8bf*)(aLoRow + ka);
        v8bf l1 = *(const v8bf*)(aLoRow + ka + 8);
        v16bf alo = __builtin_shufflevector(l0, l1,
            0,1,2,3,4,5,6,7,8,9,10,11,12,13,14,15);

        // B fragment from LDS: lane = column, lanes 0-15 K{0..15}, 16-31 K{16..31}
        #pragma unroll
        for (int t = 0; t < 4; ++t) {
            const int bofs = (t * 16 + r) * 32 + hs * 16;
            v16bf bhi = *(const v16bf*)&BsHi[buf][bofs];
            v16bf blo = *(const v16bf*)&BsLo[buf][bofs];
            acc[t] = __builtin_amdgcn_wmma_f32_16x16x32_bf16(
                false, ahi, false, bhi, (short)0, acc[t], false, false);
            acc[t] = __builtin_amdgcn_wmma_f32_16x16x32_bf16(
                false, ahi, false, blo, (short)0, acc[t], false, false);
            acc[t] = __builtin_amdgcn_wmma_f32_16x16x32_bf16(
                false, alo, false, bhi, (short)0, acc[t], false, false);
        }
        __syncthreads();                // buffer reusable next iteration
    }

    // C/D layout: VGPR i -> row = row0 + hs*8 + i, col = n0 + t*16 + (lane&15)
    #pragma unroll
    for (int t = 0; t < 4; ++t) {
        const int col = n0 + t * 16 + r;
        #pragma unroll
        for (int i = 0; i < 8; ++i)
            C[(size_t)(row0 + hs * 8 + i) * Npad + col] = acc[t][i];
    }
}

// ------------------------------------------------------------ aggregation

// AGG[i][f] = bias[f] + isd[i]^2 * G[i][f]   (self-loop term + bias init)
__global__ void agg_init_kernel(const float* __restrict__ G,
                                const float* __restrict__ bias,
                                const float* __restrict__ isd,
                                float* __restrict__ AGG,
                                int ld, int nreal) {
    size_t idx = (size_t)blockIdx.x * blockDim.x + threadIdx.x;
    size_t total = (size_t)N_NODES * ld;
    if (idx >= total) return;
    int i = (int)(idx / ld);
    int f = (int)(idx % ld);
    float sn = isd[i];
    sn *= sn;
    float b = (f < nreal) ? bias[f] : 0.0f;
    AGG[idx] = b + sn * G[idx];
}

// AGG[d][f] += isd[s]*isd[d] * G[s][f] over all edges (atomic scatter)
__global__ void agg_edges_kernel(const float* __restrict__ G,
                                 const float* __restrict__ isd,
                                 const long long* __restrict__ src,
                                 const long long* __restrict__ dst,
                                 float* __restrict__ AGG,
                                 int ld, int nreal, int nedges) {
    int e = blockIdx.x;
    if (e >= nedges) return;
    int s = (int)src[e];
    int d = (int)dst[e];
    float c = isd[s] * isd[d];
    const float* g = G + (size_t)s * ld;
    float* a = AGG + (size_t)d * ld;
    for (int f = threadIdx.x; f < nreal; f += blockDim.x)
        atomicAdd(a + f, c * g[f]);
}

// H = relu(AGG), re-split to bf16 hi/lo for GEMM2 (padded cols are already 0)
__global__ void relu_split_kernel(const float* __restrict__ AGG,
                                  __bf16* __restrict__ Hhi,
                                  __bf16* __restrict__ Hlo) {
    size_t idx = (size_t)blockIdx.x * blockDim.x + threadIdx.x;
    size_t total = (size_t)N_NODES * HID_MID_P;
    if (idx >= total) return;
    float v = fmaxf(AGG[idx], 0.0f);
    split_bf16(v, Hhi[idx], Hlo[idx]);
}

// ---------------------------------------------------------------- launcher

extern "C" void kernel_launch(void* const* d_in, const int* in_sizes, int n_in,
                              void* d_out, int out_size, void* d_ws, size_t ws_size,
                              hipStream_t stream) {
    const float*     x   = (const float*)d_in[0];
    const long long* ei  = (const long long*)d_in[1];   // int64, shape (2, E)
    const float*     W1  = (const float*)d_in[2];       // [1024, 1500]
    const float*     b1  = (const float*)d_in[3];
    const float*     W2  = (const float*)d_in[4];       // [1500, 1024]
    const float*     b2  = (const float*)d_in[5];
    const int E = in_sizes[1] / 2;
    const long long* srcE = ei;
    const long long* dstE = ei + E;

    // ---- workspace carve-up (256B aligned) ----
    char* p = (char*)d_ws;
    auto take = [&](size_t bytes) -> void* {
        void* r = (void*)p;
        p += (bytes + 255) & ~(size_t)255;
        return r;
    };
    float*  isd    = (float*)take((size_t)N_NODES * 4);
    __bf16* Xhi    = (__bf16*)take((size_t)N_NODES * HID_IN * 2);
    __bf16* Xlo    = (__bf16*)take((size_t)N_NODES * HID_IN * 2);
    __bf16* W1Thi  = (__bf16*)take((size_t)HID_MID_P * HID_IN * 2);
    __bf16* W1Tlo  = (__bf16*)take((size_t)HID_MID_P * HID_IN * 2);
    __bf16* W2Thi  = (__bf16*)take((size_t)HID_IN * HID_MID_P * 2);
    __bf16* W2Tlo  = (__bf16*)take((size_t)HID_IN * HID_MID_P * 2);
    float*  G1     = (float*)take((size_t)N_NODES * HID_MID_P * 4);
    float*  AGG1   = (float*)take((size_t)N_NODES * HID_MID_P * 4);
    __bf16* Hhi    = (__bf16*)take((size_t)N_NODES * HID_MID_P * 2);
    __bf16* Hlo    = (__bf16*)take((size_t)N_NODES * HID_MID_P * 2);
    float*  G2     = G1;   // G1 is dead after layer-1 scatter; reuse (100MB >= 64MB)
    float*  OUT    = (float*)d_out;

    const int TPB = 256;
    auto blocksFor = [](size_t n, int tpb) { return (int)((n + tpb - 1) / tpb); };

    // ---- degree / normalization ----
    deg_init_kernel<<<blocksFor(N_NODES, TPB), TPB, 0, stream>>>(isd);
    deg_count_kernel<<<blocksFor(E, TPB), TPB, 0, stream>>>(dstE, isd, E);
    deg_final_kernel<<<blocksFor(N_NODES, TPB), TPB, 0, stream>>>(isd);

    // ---- precision split of X and transposed weights ----
    split_mat_kernel<<<blocksFor((size_t)N_NODES * HID_IN, TPB), TPB, 0, stream>>>(
        x, Xhi, Xlo, (size_t)N_NODES * HID_IN);
    transpose_split_pad_kernel<<<blocksFor((size_t)HID_MID_P * HID_IN, TPB), TPB, 0, stream>>>(
        W1, W1Thi, W1Tlo, HID_IN, HID_MID, HID_MID_P, HID_IN);
    transpose_split_pad_kernel<<<blocksFor((size_t)HID_IN * HID_MID_P, TPB), TPB, 0, stream>>>(
        W2, W2Thi, W2Tlo, HID_MID, HID_IN, HID_IN, HID_MID_P);

    // ---- layer 1: G1 = X @ W1 (WMMA + async-LDS B staging) ----
    {
        int blocks = (N_NODES / 128) * (HID_MID_P / 64);   // 128 * 24 = 3072
        gemm_bf16split_lds_kernel<<<blocks, TPB, 0, stream>>>(
            Xhi, Xlo, W1Thi, W1Tlo, G1, N_NODES, HID_IN, HID_MID_P);
    }
    // ---- layer 1: aggregate + bias, relu, re-split ----
    agg_init_kernel<<<blocksFor((size_t)N_NODES * HID_MID_P, TPB), TPB, 0, stream>>>(
        G1, b1, isd, AGG1, HID_MID_P, HID_MID);
    agg_edges_kernel<<<E, TPB, 0, stream>>>(
        G1, isd, srcE, dstE, AGG1, HID_MID_P, HID_MID, E);
    relu_split_kernel<<<blocksFor((size_t)N_NODES * HID_MID_P, TPB), TPB, 0, stream>>>(
        AGG1, Hhi, Hlo);

    // ---- layer 2: G2 = H @ W2 (K padded 1500->1536 with zeros) ----
    {
        int blocks = (N_NODES / 128) * (HID_IN / 64);      // 128 * 16 = 2048
        gemm_bf16split_lds_kernel<<<blocks, TPB, 0, stream>>>(
            Hhi, Hlo, W2Thi, W2Tlo, G2, N_NODES, HID_MID_P, HID_IN);
    }
    // ---- layer 2: aggregate + bias directly into d_out ----
    agg_init_kernel<<<blocksFor((size_t)N_NODES * HID_IN, TPB), TPB, 0, stream>>>(
        G2, b2, isd, OUT, HID_IN, HID_IN);
    agg_edges_kernel<<<E, TPB, 0, stream>>>(
        G2, isd, srcE, dstE, OUT, HID_IN, HID_IN, E);
}
